// Model_5772436045968
// MI455X (gfx1250) — matrix operand order repaired and checked by reading, not compiled, not run on hardware
//
#include <hip/hip_runtime.h>
#include <stdint.h>

#define L_SEQ  128
#define BATCH  32
#define EMB    512
#define DIM    1024
#define VOCAB  32000
#define NMID   4
#define MROWS  (L_SEQ * BATCH)   // 4096

#define KBLK    64               // K elements staged per LDS tile
#define LDSROW  72               // bf16 elems per LDS row (64 + 8 pad = 144B)

typedef __attribute__((ext_vector_type(16))) __bf16 v16bf;
typedef __attribute__((ext_vector_type(8)))  float  v8f;

// GCC-style vector matching the async-to-LDS builtin's parameter type:
//   param0: v4i_vs __attribute__((address_space(1)))*   (global source)
//   param1: v4i_vs __attribute__((address_space(3)))*   (LDS destination)
typedef int v4i_vs __attribute__((vector_size(16)));
typedef __attribute__((address_space(1))) v4i_vs* gl_v4i_ptr;
typedef __attribute__((address_space(3))) v4i_vs* ld_v4i_ptr;

union FragU {
    v16bf v;
    uint4 q[2];
};

#if defined(__has_builtin)
# if __has_builtin(__builtin_amdgcn_global_load_async_to_lds_b128)
#  define HAS_ASYNC_LDS 1
# endif
#endif
#ifndef HAS_ASYNC_LDS
# define HAS_ASYNC_LDS 0
#endif

__device__ __forceinline__ unsigned short f2bf(float f) {
    unsigned u = __float_as_uint(f);
    unsigned r = u + 0x7FFFu + ((u >> 16) & 1u);   // round-to-nearest-even
    return (unsigned short)(r >> 16);
}

// ---------------------------------------------------------------------------
// fp32 -> bf16 plain convert
// ---------------------------------------------------------------------------
__global__ void cvt_bf16_kernel(const float* __restrict__ src,
                                unsigned short* __restrict__ dst, int n) {
    for (int i = blockIdx.x * blockDim.x + threadIdx.x; i < n;
         i += gridDim.x * blockDim.x)
        dst[i] = f2bf(src[i]);
}

// ---------------------------------------------------------------------------
// fp32 W(K,N) row-major -> bf16 WT(N,K) row-major
// ---------------------------------------------------------------------------
__global__ void cvt_tr_bf16_kernel(const float* __restrict__ src,
                                   unsigned short* __restrict__ dst,
                                   int K, int N) {
    int total = K * N;
    for (int i = blockIdx.x * blockDim.x + threadIdx.x; i < total;
         i += gridDim.x * blockDim.x) {
        int n = i / K;
        int k = i - n * K;
        dst[i] = f2bf(src[(size_t)k * N + n]);
    }
}

// ---------------------------------------------------------------------------
// embedding gather -> bf16 rows (MROWS, EMB)
// ---------------------------------------------------------------------------
__global__ void gather_embed_kernel(const int* __restrict__ idx,
                                    const float* __restrict__ emb,
                                    unsigned short* __restrict__ dst,
                                    int rows, int e) {
    int total = rows * e;
    for (int i = blockIdx.x * blockDim.x + threadIdx.x; i < total;
         i += gridDim.x * blockDim.x) {
        int r = i / e;
        int c = i - r * e;
        int t = idx[r];
        dst[i] = f2bf(emb[(size_t)t * e + c]);
    }
}

// ---------------------------------------------------------------------------
// WMMA bf16 GEMM:  C(M,N) f32 = A(M,K) bf16 row-major x Bt(N,K) bf16 row-major
// block = 256 threads = 8 waves; block tile = 512(M) x 64(N).
// Wave w computes 64(M) x 64(N): 4 M-subtiles x 4 N-subtiles = 16 accumulators
// -> each LDS B-fragment feeds 4 WMMAs (halves dscnt-wait frequency).
// B tile (64N x 64K) staged in LDS (double-buffered) via async DMA.
// Requires M%512==0, N%64==0, K%64==0 (true for all calls) -> EXEC all ones.
// ---------------------------------------------------------------------------
__global__ __launch_bounds__(256) void wmma_gemm_bf16(
    const unsigned short* __restrict__ A,
    const unsigned short* __restrict__ Bt,
    float* __restrict__ C,
    const float* __restrict__ bias,       // len N, may be null
    int M, int N, int K)
{
    __shared__ unsigned short ldsB[2][64 * LDSROW];   // 2 x 9216B

    const int lane = threadIdx.x & 31;
    const int wave = threadIdx.x >> 5;
    const int half = lane >> 4;          // 0: lanes 0-15, 1: lanes 16-31
    const int l15  = lane & 15;

    const int n0    = blockIdx.x * 64;
    const int mbase = blockIdx.y * 512 + wave * 64;

    // A fragment base pointers (ISA 16-bit A-matrix 16x32 layout):
    // lane holds row; elems 0..7 = K kb..kb+7, 8..15 = kb+16..kb+23, kb = k0+half*8
    const unsigned short* arow[4];
#pragma unroll
    for (int mt = 0; mt < 4; ++mt)
        arow[mt] = A + (size_t)(mbase + mt * 16 + l15) * K + half * 8;

    // cooperative stage of B tile (64 rows x 64 K-elems) into LDS buffer `buf`
    auto stage = [&](int buf, int k0) {
#pragma unroll
        for (int c = 0; c < 2; ++c) {
            int idx = threadIdx.x + c * 256;     // 512 chunks of 16B
            int row = idx >> 3;
            int seg = idx & 7;
            const unsigned short* g = Bt + (size_t)(n0 + row) * K + k0 + seg * 8;
            unsigned short* l = &ldsB[buf][row * LDSROW + seg * 8];
#if HAS_ASYNC_LDS
            __builtin_amdgcn_global_load_async_to_lds_b128(
                (gl_v4i_ptr)g, (ld_v4i_ptr)l, 0, 0);
#else
            *(uint4*)l = *(const uint4*)g;
#endif
        }
    };

    v8f acc[4][4] = {};

    stage(0, 0);

    for (int kb = 0; kb < K; kb += KBLK) {
        const int buf  = (kb / KBLK) & 1;
        const bool more = (kb + KBLK) < K;
        if (more) stage(buf ^ 1, kb + KBLK);

#if HAS_ASYNC_LDS
        // in-order async completion: <=2 outstanding means current buf landed
        if (more) {
# if __has_builtin(__builtin_amdgcn_s_wait_asynccnt)
            __builtin_amdgcn_s_wait_asynccnt(2);
# else
            asm volatile("s_wait_asynccnt 2" ::: "memory");
# endif
        } else {
# if __has_builtin(__builtin_amdgcn_s_wait_asynccnt)
            __builtin_amdgcn_s_wait_asynccnt(0);
# else
            asm volatile("s_wait_asynccnt 0" ::: "memory");
# endif
        }
#endif
        __syncthreads();   // all waves' buf is resident

#pragma unroll
        for (int mt = 0; mt < 4; ++mt)
            __builtin_prefetch(arow[mt] + kb + 2 * KBLK, 0, 3);

#pragma unroll
        for (int s = 0; s < 2; ++s) {
            const int ks = kb + s * 32;

            FragU a[4];
#pragma unroll
            for (int mt = 0; mt < 4; ++mt) {
                a[mt].q[0] = *(const uint4*)(arow[mt] + ks);
                a[mt].q[1] = *(const uint4*)(arow[mt] + ks + 16);
            }

#pragma unroll
            for (int j = 0; j < 4; ++j) {
                // B fragment: lane holds col j*16+l15; elems 0..7 = K half*8.., 8..15 = K 16+half*8..
                const unsigned short* lrow =
                    &ldsB[buf][(j * 16 + l15) * LDSROW + s * 32 + half * 8];
                FragU fb;
                fb.q[0] = *(const uint4*)(lrow);
                fb.q[1] = *(const uint4*)(lrow + 16);
#pragma unroll
                for (int mt = 0; mt < 4; ++mt)
                    acc[mt][j] = __builtin_amdgcn_wmma_f32_16x16x32_bf16(
                        false, a[mt].v, false, fb.v, (short)0, acc[mt][j],
                        false, false);
            }
        }
        __syncthreads();   // protect buf from being overwritten next iter
    }

    // C/D layout: VGPR v -> row mt*16 + half*8 + v, col n0 + 16j + l15
    const int col = n0 + l15;
    float bv[4] = {0.f, 0.f, 0.f, 0.f};
    if (bias) {
#pragma unroll
        for (int j = 0; j < 4; ++j) bv[j] = bias[col + 16 * j];
    }
#pragma unroll
    for (int mt = 0; mt < 4; ++mt) {
        const int row = mbase + mt * 16 + half * 8;
#pragma unroll
        for (int v = 0; v < 8; ++v) {
            float* crow = C + (size_t)(row + v) * N;
#pragma unroll
            for (int j = 0; j < 4; ++j)
                crow[col + 16 * j] = acc[mt][j][v] + bv[j];
        }
    }
}

// ---------------------------------------------------------------------------
// SRU recurrence: one thread per (b, j); serial over L.
// U: (MROWS, k*n).  hw = U[...,3n:] if k==4 else xin.  Writes f32 + bf16 H.
// ---------------------------------------------------------------------------
__global__ void sru_scan_kernel(const float* __restrict__ U,
                                const float* __restrict__ xin,
                                const float* __restrict__ bias,
                                const float* __restrict__ c0,
                                float* __restrict__ Hout,
                                unsigned short* __restrict__ HoutB,
                                int n, int k)
{
    int t = blockIdx.x * blockDim.x + threadIdx.x;
    if (t >= BATCH * n) return;
    int b = t / n;
    int j = t - b * n;

    float c  = c0[t];
    float bf = bias[j];
    float br = bias[n + j];
    const int kn = k * n;

    for (int l = 0; l < L_SEQ; ++l) {
        size_t row = (size_t)(l * BATCH + b);
        const float* u = U + row * kn;
        float xt = u[j];
        float f  = 1.f / (1.f + __expf(-(u[n + j]     + bf)));
        float r  = 1.f / (1.f + __expf(-(u[2 * n + j] + br)));
        float hw = (k == 4) ? u[3 * n + j] : xin[row * n + j];
        c = f * c + (1.f - f) * xt;
        float g = tanhf(c);
        float h = r * g + (1.f - r) * hw;
        Hout[row * n + j]  = h;
        HoutB[row * n + j] = f2bf(h);
    }
}

// ---------------------------------------------------------------------------
extern "C" void kernel_launch(void* const* d_in, const int* in_sizes, int n_in,
                              void* d_out, int out_size, void* d_ws, size_t ws_size,
                              hipStream_t stream) {
    const int*   x        = (const int*)d_in[0];
    const float* c1       = (const float*)d_in[1];
    const float* c2       = (const float*)d_in[2];
    const float* c3       = (const float*)d_in[3];
    const float* emb      = (const float*)d_in[4];
    const float* out_bias = (const float*)d_in[5];
    const float* W1       = (const float*)d_in[6];
    const float* b1       = (const float*)d_in[7];
    const float* W2       = (const float*)d_in[8];
    const float* b2       = (const float*)d_in[9];
    const float* W3       = (const float*)d_in[10];
    const float* b3       = (const float*)d_in[11];
    float* out = (float*)d_out;

    char* wsp = (char*)d_ws;
    size_t off = 0;
    auto alloc = [&](size_t bytes) -> void* {
        void* p = wsp + off;
        off = (off + bytes + 255) & ~(size_t)255;
        return p;
    };

    unsigned short* embB = (unsigned short*)alloc((size_t)VOCAB * EMB * 2);
    unsigned short* W1T  = (unsigned short*)alloc((size_t)4 * DIM * EMB * 2);
    unsigned short* W2T  = (unsigned short*)alloc((size_t)NMID * 3 * DIM * DIM * 2);
    unsigned short* W3T  = (unsigned short*)alloc((size_t)4 * EMB * DIM * 2);
    unsigned short* XB   = (unsigned short*)alloc((size_t)MROWS * DIM * 2);
    float* U  = (float*)alloc((size_t)MROWS * 4 * DIM * 4);
    float* H0 = (float*)alloc((size_t)MROWS * DIM * 4);
    float* H1 = (float*)alloc((size_t)MROWS * DIM * 4);

    // weight / embedding conversion (bf16 weights all fit in 192MB L2)
    cvt_bf16_kernel<<<2048, 256, 0, stream>>>(emb, embB, VOCAB * EMB);
    cvt_tr_bf16_kernel<<<2048, 256, 0, stream>>>(W1, W1T, EMB, 4 * DIM);
    for (int i = 0; i < NMID; ++i)
        cvt_tr_bf16_kernel<<<2048, 256, 0, stream>>>(
            W2 + (size_t)i * DIM * 3 * DIM, W2T + (size_t)i * 3 * DIM * DIM,
            DIM, 3 * DIM);
    cvt_tr_bf16_kernel<<<2048, 256, 0, stream>>>(W3, W3T, DIM, 4 * EMB);

    gather_embed_kernel<<<2048, 256, 0, stream>>>(x, emb, XB, MROWS, EMB);

    // layer 1: E -> D, k=4
    wmma_gemm_bf16<<<dim3(4 * DIM / 64, MROWS / 512), 256, 0, stream>>>(
        XB, W1T, U, nullptr, MROWS, 4 * DIM, EMB);
    sru_scan_kernel<<<(BATCH * DIM + 255) / 256, 256, 0, stream>>>(
        U, nullptr, b1, c1, H0, XB, DIM, 4);

    // middle layers: D -> D, k=3 (hw = layer input)
    float* hprev = H0;
    float* hcur  = H1;
    for (int i = 0; i < NMID; ++i) {
        wmma_gemm_bf16<<<dim3(3 * DIM / 64, MROWS / 512), 256, 0, stream>>>(
            XB, W2T + (size_t)i * 3 * DIM * DIM, U, nullptr, MROWS, 3 * DIM, DIM);
        sru_scan_kernel<<<(BATCH * DIM + 255) / 256, 256, 0, stream>>>(
            U, hprev, b2 + (size_t)i * 2 * DIM, c2 + (size_t)i * BATCH * DIM,
            hcur, XB, DIM, 3);
        float* tmp = hprev; hprev = hcur; hcur = tmp;
    }

    // layer 3: D -> E, k=4
    wmma_gemm_bf16<<<dim3(4 * EMB / 64, MROWS / 512), 256, 0, stream>>>(
        XB, W3T, U, nullptr, MROWS, 4 * EMB, DIM);
    sru_scan_kernel<<<(BATCH * EMB + 255) / 256, 256, 0, stream>>>(
        U, nullptr, b3, c3, hcur, XB, EMB, 4);

    // vocab projection: logits = H(4096,512) @ emb^T + out_bias
    // embB (V,E) row-major is exactly Bt(N=V, K=E)
    wmma_gemm_bf16<<<dim3(VOCAB / 64, MROWS / 512), 256, 0, stream>>>(
        XB, embB, out, out_bias, MROWS, VOCAB, EMB);
}
